// ExBiMamba_35631048687798
// MI455X (gfx1250) — compile-verified, matrix operand order checked by reading, not run on hardware
//
#include <hip/hip_runtime.h>
#include <hip/hip_bf16.h>
#include <math.h>

// ---------------- Problem constants (fixed by the reference) ----------------
#define D_MODEL  1024
#define D_STATE  16
#define D_CONV   4
#define D_INNER  2048
#define DT_RANK  64
#define BATCH    2
#define SEQ      1024
#define MROWS    (BATCH*SEQ)     // 2048 token rows

// ---------------- Vector types ----------------
typedef __attribute__((ext_vector_type(16))) __bf16 v16bf;
typedef __attribute__((ext_vector_type(8)))  float  v8f;
typedef __attribute__((ext_vector_type(4)))  float  v4f;
typedef int v4i_vs __attribute__((vector_size(4 * sizeof(int))));

// ---------------- gfx1250 async global->LDS copy (ASYNCcnt path) ----------------
#if defined(__has_builtin)
#  if __has_builtin(__builtin_amdgcn_global_load_async_to_lds_b128)
#    define HAVE_ASYNC_LDS 1
#  endif
#  if __has_builtin(__builtin_amdgcn_s_wait_asynccnt)
#    define HAVE_WAIT_ASYNC 1
#  endif
#endif
#ifndef HAVE_ASYNC_LDS
#  define HAVE_ASYNC_LDS 0
#endif
#ifndef HAVE_WAIT_ASYNC
#  define HAVE_WAIT_ASYNC 0
#endif

template <int N>
__device__ __forceinline__ void wait_asynccnt() {
#if HAVE_ASYNC_LDS
#  if HAVE_WAIT_ASYNC
    __builtin_amdgcn_s_wait_asynccnt(N);
#  else
    asm volatile("s_wait_asynccnt %0" :: "i"(N) : "memory");
#  endif
#endif
}

__device__ __forceinline__ void async_copy16(const float* g, float* l) {
#if HAVE_ASYNC_LDS
    __builtin_amdgcn_global_load_async_to_lds_b128(
        (__attribute__((address_space(1))) v4i_vs*)g,
        (__attribute__((address_space(3))) v4i_vs*)l, 0, 0);
#else
    *(v4f*)l = *(const v4f*)g;       // fallback: register-staged copy
#endif
}

__device__ __forceinline__ float softplusf(float x) {
    return (x > 20.f) ? x : log1pf(__expf(x));
}
__device__ __forceinline__ float siluf(float x) {
    return x / (1.f + __expf(-x));
}

// =====================================================================
// WMMA-bf16 GEMM:  C[M,N] = epilogue( A[M,K] @ W[N,K]^T )
//   BM=BN=128, BK=32; 256 threads = 8 wave32 (2x4 wave grid).
//   Double-buffered fp32 LDS tiles filled by async global->LDS DMA;
//   fp32 -> bf16 conversion happens at fragment-load time (co-executes
//   with WMMA). Padded LDS rows (36 floats = 144B) keep the b128
//   fragment reads bank-conflict-free.
//   mode 0: plain store; 1: softplus(acc+bias); 2: acc+bias
//   flip_a / flip_c: time-axis flip on input rows / output rows
// =====================================================================
__launch_bounds__(256)
__global__ void gemm_wmma_bf16(const float* __restrict__ A, int lda,
                               const float* __restrict__ W, int ldw,
                               float* __restrict__ C, int ldc,
                               int N, int K,
                               const float* __restrict__ bias,
                               int mode, int col_off, int flip_a, int flip_c)
{
    constexpr int BM = 128, BN = 128, BK = 32, LDP = 36;
    __shared__ __align__(16) float As[2][BM * LDP];
    __shared__ __align__(16) float Ws[2][BN * LDP];

    const int tid  = threadIdx.x;
    const int wave = tid >> 5;
    const int lane = tid & 31;
    const int half = lane >> 4;          // K-half selector in WMMA fragments
    const int lr   = lane & 15;

    const int bm = blockIdx.y * BM;
    const int bn = blockIdx.x * BN;
    const int wm = (wave >> 2) * 64;     // 2 waves along M (4 m-tiles each)
    const int wn = (wave & 3) * 32;      // 4 waves along N (2 n-tiles each)

    // staging coordinates: 4 passes, each thread moves one b128 (4 floats)
    const int src_r  = tid >> 3;         // +32 per pass
    const int src_c4 = (tid & 7) * 4;

    // issue one K-tile (8 async-copy instructions per wave: 4 A + 4 W)
    auto stage = [&](int k0, int buf) {
        #pragma unroll
        for (int p = 0; p < 4; ++p) {
            int r = p * 32 + src_r;
            int grow = bm + r;
            if (flip_a) { int b = grow >> 10; int t = grow & (SEQ - 1); grow = (b << 10) + (SEQ - 1 - t); }
            async_copy16(A + (size_t)grow * lda + k0 + src_c4, &As[buf][r * LDP + src_c4]);
        }
        #pragma unroll
        for (int p = 0; p < 4; ++p) {
            int r = p * 32 + src_r;
            int gn = bn + r; if (gn > N - 1) gn = N - 1;   // clamp; garbage cols never stored
            async_copy16(W + (size_t)gn * ldw + k0 + src_c4, &Ws[buf][r * LDP + src_c4]);
        }
    };

    // load one 16x32 bf16 fragment from an fp32 LDS tile (ISA 16-bit A/B layout)
    auto frag = [&](const float* base, int row) -> v16bf {
        const float* rowp = base + row * LDP;
        v4f q0 = *(const v4f*)(rowp + half * 8);           // K = h*8 .. h*8+3
        v4f q1 = *(const v4f*)(rowp + half * 8 + 4);       // K = h*8+4 .. h*8+7
        v4f q2 = *(const v4f*)(rowp + half * 8 + 16);      // K = 16+h*8 ..
        v4f q3 = *(const v4f*)(rowp + half * 8 + 20);
        v16bf f;
        f[0]  = (__bf16)q0.x; f[1]  = (__bf16)q0.y; f[2]  = (__bf16)q0.z; f[3]  = (__bf16)q0.w;
        f[4]  = (__bf16)q1.x; f[5]  = (__bf16)q1.y; f[6]  = (__bf16)q1.z; f[7]  = (__bf16)q1.w;
        f[8]  = (__bf16)q2.x; f[9]  = (__bf16)q2.y; f[10] = (__bf16)q2.z; f[11] = (__bf16)q2.w;
        f[12] = (__bf16)q3.x; f[13] = (__bf16)q3.y; f[14] = (__bf16)q3.z; f[15] = (__bf16)q3.w;
        return f;
    };

    v8f acc[4][2];
    #pragma unroll
    for (int mt = 0; mt < 4; ++mt)
        #pragma unroll
        for (int nt = 0; nt < 2; ++nt)
            acc[mt][nt] = (v8f){0.f,0.f,0.f,0.f,0.f,0.f,0.f,0.f};

    const int nk = K / BK;
    stage(0, 0);
    int cur = 0;

    for (int kk = 0; kk < nk; ++kk) {
        if (kk + 1 < nk) {
            stage((kk + 1) * BK, cur ^ 1);    // prefetch next tile into other buffer
            wait_asynccnt<8>();               // the 8 older copies (tile kk) are done
        } else {
            wait_asynccnt<0>();
        }
        __syncthreads();                      // all waves' parts of tile kk landed

        v16bf af[4], bf[2];
        #pragma unroll
        for (int mt = 0; mt < 4; ++mt) af[mt] = frag(As[cur], wm + mt * 16 + lr);
        #pragma unroll
        for (int nt = 0; nt < 2; ++nt) bf[nt] = frag(Ws[cur], wn + nt * 16 + lr);
        #pragma unroll
        for (int mt = 0; mt < 4; ++mt)
            #pragma unroll
            for (int nt = 0; nt < 2; ++nt)
                acc[mt][nt] = __builtin_amdgcn_wmma_f32_16x16x32_bf16(
                    false, af[mt], false, bf[nt], (short)0, acc[mt][nt], false, false);

        __syncthreads();                      // reads done before buffer is overwritten
        cur ^= 1;
    }

    // ---- epilogue: D layout = VGPR i, lanes 0-15: M=i,N=lane; 16-31: M=8+i ----
    #pragma unroll
    for (int nt = 0; nt < 2; ++nt) {
        int col = bn + wn + nt * 16 + lr;
        if (col >= N) continue;
        float bv = (mode != 0) ? bias[col] : 0.f;
        #pragma unroll
        for (int mt = 0; mt < 4; ++mt) {
            #pragma unroll
            for (int i = 0; i < 8; ++i) {
                int row = bm + wm + mt * 16 + half * 8 + i;
                float v = acc[mt][nt][i];
                if (mode == 1) v = softplusf(v + bv);
                else if (mode == 2) v += bv;
                int orow = row;
                if (flip_c) { int b = row >> 10; int t = row & (SEQ - 1); orow = (b << 10) + (SEQ - 1 - t); }
                C[(size_t)orow * ldc + col + col_off] = v;
            }
        }
    }
}

// =====================================================================
// Causal depthwise conv (k=4) + bias + SiLU.  xi lives in cols [0,2048) of xz.
// =====================================================================
__global__ void conv_silu_kernel(const float* __restrict__ xz,      // (MROWS, 4096)
                                 const float* __restrict__ conv_w,  // (D_INNER, 4)
                                 const float* __restrict__ conv_b,  // (D_INNER)
                                 float* __restrict__ xi_out)        // (MROWS, D_INNER)
{
    int idx = blockIdx.x * blockDim.x + threadIdx.x;
    if (idx >= MROWS * D_INNER) return;
    int d   = idx & (D_INNER - 1);
    int row = idx >> 11;
    int t   = row & (SEQ - 1);
    int bb  = row >> 10;
    float s = conv_b[d];
    #pragma unroll
    for (int j = 0; j < D_CONV; ++j) {
        int tt = t - (D_CONV - 1) + j;
        if (tt >= 0)
            s += conv_w[d * D_CONV + j] * xz[(size_t)((bb << 10) + tt) * 4096 + d];
    }
    xi_out[idx] = siluf(s);
}

// =====================================================================
// Selective scan + gating. One lane per (channel d, state n); 16-lane
// shfl_xor butterfly reduces the state dimension.
// grid.x = BATCH * D_INNER / 16 = 256 blocks of 256 threads.
// =====================================================================
__launch_bounds__(256)
__global__ void scan_kernel(const float* __restrict__ delta,  // (MROWS, D_INNER)
                            const float* __restrict__ u,      // conv(SiLU) (MROWS, D_INNER)
                            const float* __restrict__ xz,     // z at col 2048+d (MROWS, 4096)
                            const float* __restrict__ dbl,    // (MROWS, 96): B at 64+n, C at 80+n
                            const float* __restrict__ A_log,  // (D_INNER, 16)
                            const float* __restrict__ Dp,     // (D_INNER)
                            float* __restrict__ yg)           // (MROWS, D_INNER)
{
    int tid  = threadIdx.x;
    int n    = tid & 15;
    int chan = blockIdx.x * 16 + (tid >> 4);  // 0 .. BATCH*D_INNER-1
    int d    = chan & (D_INNER - 1);
    int bb   = chan >> 11;

    float An = -__expf(A_log[d * D_STATE + n]);
    float Dd = Dp[d];
    float h  = 0.f;
    const size_t base = (size_t)bb * SEQ;

    for (int t = 0; t < SEQ; ++t) {
        size_t row = base + t;
        float dv = delta[row * D_INNER + d];
        float uv = u[row * D_INNER + d];
        float Bt = dbl[row * 96 + DT_RANK + n];
        float Ct = dbl[row * 96 + DT_RANK + D_STATE + n];
        float dA = __expf(dv * An);
        h = h * dA + (dv * uv) * Bt;
        float yp = h * Ct;
        yp += __shfl_xor(yp, 1, 16);
        yp += __shfl_xor(yp, 2, 16);
        yp += __shfl_xor(yp, 4, 16);
        yp += __shfl_xor(yp, 8, 16);
        if (n == 0) {
            float z = xz[row * 4096 + D_INNER + d];
            yg[row * D_INNER + d] = (yp + uv * Dd) * siluf(z);
        }
    }
}

// =====================================================================
// Orchestration
// =====================================================================
extern "C" void kernel_launch(void* const* d_in, const int* in_sizes, int n_in,
                              void* d_out, int out_size, void* d_ws, size_t ws_size,
                              hipStream_t stream) {
    // Input order: x, params_f{in_w,conv_w,conv_b,x_w,dt_w,dt_b,A_log,D,out_w},
    //              params_b{...}, proj_w, proj_b
    const float* x        = (const float*)d_in[0];
    const float* in_w[2]   = { (const float*)d_in[1],  (const float*)d_in[10] };
    const float* conv_w[2] = { (const float*)d_in[2],  (const float*)d_in[11] };
    const float* conv_b[2] = { (const float*)d_in[3],  (const float*)d_in[12] };
    const float* x_w[2]    = { (const float*)d_in[4],  (const float*)d_in[13] };
    const float* dt_w[2]   = { (const float*)d_in[5],  (const float*)d_in[14] };
    const float* dt_b[2]   = { (const float*)d_in[6],  (const float*)d_in[15] };
    const float* A_log[2]  = { (const float*)d_in[7],  (const float*)d_in[16] };
    const float* Dp[2]     = { (const float*)d_in[8],  (const float*)d_in[17] };
    const float* out_w[2]  = { (const float*)d_in[9],  (const float*)d_in[18] };
    const float* proj_w   = (const float*)d_in[19];
    const float* proj_b   = (const float*)d_in[20];
    float* out = (float*)d_out;

    // Workspace layout (floats)
    float* ws = (float*)d_ws;
    size_t o = 0;
    float* xz[2];  xz[0]  = ws + o; o += (size_t)MROWS * 4096;    xz[1]  = ws + o; o += (size_t)MROWS * 4096;
    float* xi[2];  xi[0]  = ws + o; o += (size_t)MROWS * D_INNER; xi[1]  = ws + o; o += (size_t)MROWS * D_INNER;
    float* dbl[2]; dbl[0] = ws + o; o += (size_t)MROWS * 96;      dbl[1] = ws + o; o += (size_t)MROWS * 96;
    float* dlt[2]; dlt[0] = ws + o; o += (size_t)MROWS * D_INNER; dlt[1] = ws + o; o += (size_t)MROWS * D_INNER;
    float* yg[2];  yg[0]  = ws + o; o += (size_t)MROWS * D_INNER; yg[1]  = ws + o; o += (size_t)MROWS * D_INNER;
    float* ycat = ws + o; o += (size_t)MROWS * (2 * D_MODEL);

    dim3 blk(256);
    const int GM = MROWS / 128;  // 16

    for (int dir = 0; dir < 2; ++dir) {
        int flip = dir;  // backward direction runs in flipped time order
        // in_proj: xz = x @ in_w^T  (N=4096, K=1024); flip A rows for backward
        gemm_wmma_bf16<<<dim3(4096 / 128, GM), blk, 0, stream>>>(
            x, D_MODEL, in_w[dir], D_MODEL, xz[dir], 4096,
            4096, D_MODEL, nullptr, 0, 0, flip, 0);
        // conv + silu on xi half
        conv_silu_kernel<<<dim3((MROWS * D_INNER) / 256), blk, 0, stream>>>(
            xz[dir], conv_w[dir], conv_b[dir], xi[dir]);
        // x_proj: dbl = xi @ x_w^T  (N=96, K=2048)
        gemm_wmma_bf16<<<dim3(1, GM), blk, 0, stream>>>(
            xi[dir], D_INNER, x_w[dir], D_INNER, dbl[dir], 96,
            96, D_INNER, nullptr, 0, 0, 0, 0);
        // dt_proj + softplus: delta = softplus(dbl[:, :64] @ dt_w^T + dt_b)
        gemm_wmma_bf16<<<dim3(D_INNER / 128, GM), blk, 0, stream>>>(
            dbl[dir], 96, dt_w[dir], DT_RANK, dlt[dir], D_INNER,
            D_INNER, DT_RANK, dt_b[dir], 1, 0, 0, 0);
        // selective scan + gating
        scan_kernel<<<dim3(BATCH * D_INNER / 16), blk, 0, stream>>>(
            dlt[dir], xi[dir], xz[dir], dbl[dir], A_log[dir], Dp[dir], yg[dir]);
        // out_proj into concat buffer; un-flip backward rows on the way out
        gemm_wmma_bf16<<<dim3(D_MODEL / 128, GM), blk, 0, stream>>>(
            yg[dir], D_INNER, out_w[dir], D_INNER, ycat, 2 * D_MODEL,
            D_MODEL, D_INNER, nullptr, 0, dir * D_MODEL, 0, flip);
    }
    // final projection: out = ycat @ proj_w^T + proj_b  (N=1024, K=2048)
    gemm_wmma_bf16<<<dim3(D_MODEL / 128, GM), blk, 0, stream>>>(
        ycat, 2 * D_MODEL, proj_w, 2 * D_MODEL, out, D_MODEL,
        D_MODEL, 2 * D_MODEL, proj_b, 2, 0, 0, 0);
}
